// DeepFM_67954972557815
// MI455X (gfx1250) — compile-verified, matrix operand order checked by reading, not run on hardware
//
#include <hip/hip_runtime.h>
#include <hip/hip_bf16.h>

#define FIELD 39
#define VOCAB 100000
#define EMB   16
#define BATCH 16384
#define D1    400
#define D2    400

#define KDEEP   (FIELD*EMB)   // 624
#define K1      640           // 624 padded to multiple of 32
#define K1S     648           // LDS row stride (bank spread, keeps 16B align)
#define K2      416           // 400 padded to multiple of 32
#define K2S     424           // LDS row stride
#define TILE_B  16
#define NT1     (D1/16)       // 25 N-tiles
#define NT2     (D2/16)       // 25 N-tiles

#define STAGE_BYTES (TILE_B*FIELD*EMB*4)   // 39936 B f32 staging (>= H bytes)

typedef __bf16 bf16_t;
typedef __attribute__((ext_vector_type(16))) __bf16 v16bf;
typedef __attribute__((ext_vector_type(8)))  __bf16 v8bf;
typedef __attribute__((ext_vector_type(8)))  float  v8f;
typedef __attribute__((ext_vector_type(4)))  float  v4f;

// ---------------------------------------------------------------------------
// Weight prep: transpose W1/W2 to K-major bf16 with zero-padded K so each
// WMMA B-fragment is two contiguous 16-byte loads per lane.
// ---------------------------------------------------------------------------
__global__ void deepfm_prep_weights(const float* __restrict__ W1,
                                    const float* __restrict__ W2,
                                    bf16_t* __restrict__ w1t,
                                    bf16_t* __restrict__ w2t) {
    const int k = blockIdx.x;
    if (blockIdx.y == 0) {
        if (k < K1) {
            for (int n = threadIdx.x; n < D1; n += blockDim.x) {
                float v = (k < KDEEP) ? W1[(size_t)k * D1 + n] : 0.0f;
                w1t[(size_t)n * K1 + k] = (bf16_t)v;
            }
        }
    } else {
        if (k < K2) {
            for (int n = threadIdx.x; n < D2; n += blockDim.x) {
                float v = (k < D1) ? W2[(size_t)k * D2 + n] : 0.0f;
                w2t[(size_t)n * K2 + k] = (bf16_t)v;
            }
        }
    }
}

// issue one 16B async global->LDS copy (tracked by ASYNCcnt, no VGPR data)
__device__ __forceinline__ void async_copy_b128(const void* gptr, void* lptr) {
    unsigned int  lds  = (unsigned int)(unsigned long long)lptr; // LDS byte offset
    unsigned long long ga = (unsigned long long)gptr;
    asm volatile("global_load_async_to_lds_b128 %0, %1, off"
                 :: "v"(lds), "v"(ga) : "memory");
}
__device__ __forceinline__ void async_wait_all() {
    asm volatile("s_wait_asynccnt 0x0" ::: "memory");
}

// ---------------------------------------------------------------------------
// Fused DeepFM forward for a 16-row batch tile. 128 threads = 4 waves.
// ---------------------------------------------------------------------------
__global__ __launch_bounds__(128)
void deepfm_fused(const int*    __restrict__ Xi,
                  const float*  __restrict__ Xv,
                  const float*  __restrict__ first_table,
                  const float*  __restrict__ second_table,
                  const bf16_t* __restrict__ w1t,
                  const float*  __restrict__ b1,
                  const bf16_t* __restrict__ w2t,
                  const float*  __restrict__ b2,
                  const float*  __restrict__ bias,
                  float*        __restrict__ out) {
    __shared__ bf16_t A[TILE_B * K1S];               // bf16 deep_emb tile
    __shared__ __align__(16) unsigned char shmix[STAGE_BYTES];
    float*  stage = (float*)shmix;                   // f32 gather staging
    bf16_t* H     = (bf16_t*)shmix;                  // overlaid: hidden tile
    __shared__ float fm_acc[TILE_B];
    __shared__ float rowsum[TILE_B];

    const int tid    = threadIdx.x;
    const int bstart = blockIdx.x * TILE_B;

    if (tid < TILE_B) { fm_acc[tid] = 0.0f; rowsum[tid] = 0.0f; }
    for (int i = tid; i < TILE_B * 16; i += 128) {   // A K-pad: k in [624,640)
        int r = i >> 4, kk = i & 15;
        A[r * K1S + KDEEP + kk] = (bf16_t)0.0f;
    }
    __syncthreads();

    // ---- Stage 0: async gathers into LDS staging + first-order FM ----------
    // 4 lanes cooperate per (row,field): each copies 16B of the 64B emb row.
    {
        const int j = tid & 3;                       // 16B chunk index
        for (int p = (tid >> 2); p < TILE_B * FIELD; p += 32) {
            const int r = p / FIELD, f = p % FIELD;
            const int b = bstart + r;
            const int idx = Xi[(size_t)b * FIELD + f];
            const float* src = second_table + ((size_t)f * VOCAB + idx) * EMB + j * 4;
            async_copy_b128(src, &stage[(size_t)p * EMB + j * 4]);
            if (j == 0) {
                const float xv = Xv[(size_t)b * FIELD + f];
                const float w  = first_table[(size_t)f * VOCAB + idx];
                atomicAdd(&fm_acc[r], w * xv);
            }
        }
    }
    async_wait_all();
    __syncthreads();

    // ---- Stage 1: scale by Xv, convert to bf16 WMMA A-tile -----------------
    for (int p = tid; p < TILE_B * FIELD; p += 128) {
        const int r = p / FIELD, f = p % FIELD;
        const float xv = Xv[(size_t)(bstart + r) * FIELD + f];
        const float* s = &stage[(size_t)p * EMB];
        bf16_t* dst = &A[r * K1S + f * EMB];
#pragma unroll
        for (int jj = 0; jj < EMB; ++jj) dst[jj] = (bf16_t)(s[jj] * xv);
    }
    __syncthreads();                                  // staging dead after this

    // H K-pad: k in [400,416)  (H overlays the now-dead staging buffer)
    for (int i = tid; i < TILE_B * 16; i += 128) {
        int r = i >> 4, kk = i & 15;
        H[r * K2S + D1 + kk] = (bf16_t)0.0f;
    }

    // ---- Stage 2: second-order FM from the LDS A-tile ----------------------
    {
        const int r  = tid >> 3;                      // 0..15
        const int e0 = (tid & 7) * 2;                 // two emb dims per thread
        float acc = 0.0f;
#pragma unroll
        for (int ei = 0; ei < 2; ++ei) {
            const int e = e0 + ei;
            float s = 0.0f, ss = 0.0f;
            for (int f = 0; f < FIELD; ++f) {
                float v = (float)A[r * K1S + f * EMB + e];
                s += v; ss += v * v;
            }
            acc += 0.5f * (s * s - ss);
        }
        atomicAdd(&fm_acc[r], acc);
    }

    const int wave = tid >> 5;
    const int lane = tid & 31;
    const int nlo  = lane & 15;          // N within tile / A-row M
    const int kh   = lane >> 4;          // K-half select per WMMA layout

    // ---- Stage 3: GEMM1  H = relu(A[16xK1] * W1 + b1) ----------------------
    for (int nt = wave; nt < NT1; nt += 4) {
        const int ncol = nt * 16 + nlo;
        const float bv = b1[ncol];
        v8f acc = { bv, bv, bv, bv, bv, bv, bv, bv };
        const bf16_t* brow = w1t + (size_t)ncol * K1;
        if (nt + 4 < NT1)                 // pull next B panel toward L0/L2
            __builtin_prefetch(w1t + (size_t)(ncol + 64) * K1, 0, 0);
        for (int ks = 0; ks < K1 / 32; ++ks) {
            const int k0 = ks * 32 + kh * 8;
            union { v16bf v; v8bf h[2]; } ua, ub;
            ua.h[0] = *(const v8bf*)(&A[nlo * K1S + k0]);
            ua.h[1] = *(const v8bf*)(&A[nlo * K1S + k0 + 16]);
            ub.h[0] = *(const v8bf*)(brow + k0);
            ub.h[1] = *(const v8bf*)(brow + k0 + 16);
            acc = __builtin_amdgcn_wmma_f32_16x16x32_bf16(
                false, ua.v, false, ub.v, (short)0, acc, false, false);
        }
#pragma unroll
        for (int r = 0; r < 8; ++r) {
            float v = acc[r] < 0.0f ? 0.0f : acc[r];
            const int m = r + 8 * kh;          // C layout: M = r + 8*(lane>=16)
            H[m * K2S + ncol] = (bf16_t)v;
        }
    }
    __syncthreads();

    // ---- Stage 4: GEMM2 + relu + per-row sum -------------------------------
    for (int nt = wave; nt < NT2; nt += 4) {
        const int ncol = nt * 16 + nlo;
        const float bv = b2[ncol];
        v8f acc = { bv, bv, bv, bv, bv, bv, bv, bv };
        const bf16_t* brow = w2t + (size_t)ncol * K2;
        if (nt + 4 < NT2)
            __builtin_prefetch(w2t + (size_t)(ncol + 64) * K2, 0, 0);
        for (int ks = 0; ks < K2 / 32; ++ks) {
            const int k0 = ks * 32 + kh * 8;
            union { v16bf v; v8bf h[2]; } ua, ub;
            ua.h[0] = *(const v8bf*)(&H[nlo * K2S + k0]);
            ua.h[1] = *(const v8bf*)(&H[nlo * K2S + k0 + 16]);
            ub.h[0] = *(const v8bf*)(brow + k0);
            ub.h[1] = *(const v8bf*)(brow + k0 + 16);
            acc = __builtin_amdgcn_wmma_f32_16x16x32_bf16(
                false, ua.v, false, ub.v, (short)0, acc, false, false);
        }
#pragma unroll
        for (int r = 0; r < 8; ++r) {
            float v = acc[r] < 0.0f ? 0.0f : acc[r];
            v += __shfl_xor(v, 1, 32);
            v += __shfl_xor(v, 2, 32);
            v += __shfl_xor(v, 4, 32);
            v += __shfl_xor(v, 8, 32);
            if (nlo == 0) atomicAdd(&rowsum[r + 8 * kh], v);
        }
    }
    __syncthreads();

    // ---- Stage 5: final combine -------------------------------------------
    if (tid < TILE_B) {
        out[bstart + tid] = fm_acc[tid] + rowsum[tid] + bias[0];
    }
}

extern "C" void kernel_launch(void* const* d_in, const int* in_sizes, int n_in,
                              void* d_out, int out_size, void* d_ws, size_t ws_size,
                              hipStream_t stream) {
    const int*   Xi   = (const int*)  d_in[0];
    const float* Xv   = (const float*)d_in[1];
    const float* ft   = (const float*)d_in[2];
    const float* st   = (const float*)d_in[3];
    const float* W1   = (const float*)d_in[4];
    const float* b1   = (const float*)d_in[5];
    const float* W2   = (const float*)d_in[6];
    const float* b2   = (const float*)d_in[7];
    const float* bias = (const float*)d_in[8];
    float* out = (float*)d_out;

    bf16_t* w1t = (bf16_t*)d_ws;                                   // D1*K1 bf16
    bf16_t* w2t = (bf16_t*)((char*)d_ws + (size_t)D1 * K1 * 2);    // D2*K2 bf16

    deepfm_prep_weights<<<dim3(K1, 2), 128, 0, stream>>>(W1, W2, w1t, w2t);
    deepfm_fused<<<BATCH / TILE_B, 128, 0, stream>>>(Xi, Xv, ft, st,
                                                     w1t, b1, w2t, b2, bias, out);
}
